// SpatialFeatureExtractor_2001454760525
// MI455X (gfx1250) — matrix operand order repaired and checked by reading, not compiled, not run on hardware
//
#include <hip/hip_runtime.h>
#include <hip/hip_bf16.h>

// ---------------------------------------------------------------------------
// Fused Conv1d(1->128,k3)+BN+ReLU -> Conv1d(128->256)+BN+ReLU ->
// Conv1d(256->256)+BN+ReLU -> mean over L, for 1024 independent rows of 520.
// One row per workgroup; activations staged in LDS as bf16; conv layers 2/3
// run as im2col GEMMs on v_wmma_f32_16x16x32_bf16.
// ---------------------------------------------------------------------------

typedef __attribute__((ext_vector_type(16))) __bf16       v16bf;
typedef __attribute__((ext_vector_type(8)))  float        v8f;
typedef __attribute__((ext_vector_type(4)))  unsigned int v4u;
typedef __attribute__((ext_vector_type(8)))  unsigned int v8u;

#define L_TOTAL 520
#define CHUNK   104      // 520 = 5 * 104
#define NCHUNK  5
#define NT      7        // 7 n-tiles * 16 = 112 local positions (covers CHUNK+2 halo)
#define H1ROWS  114      // rows j: position = base-2+j ; reads go up to j2+2 = 113
#define H1S     136      // halves per row: 128 + 8 pad (bank stagger)
#define H2ROWS  114
#define H2S     264      // 256 + 8 pad
#define C1      128
#define C3      256
#define KB2     12       // K = 384 / 32
#define KB3     24       // K = 768 / 32

// ---------------------------------------------------------------------------
// Repack w2 [256][128][3] and w3 [256][256][3] (f32) into bf16 WMMA A-matrix
// fragments. K ordering: K = ci + Cin*k. Per CDNA5 16-bit A 16x32 layout:
//   lane<16 : halves t=0..7 -> K0+t,   t=8..15 -> K0+t+8   (K 0..7,16..23)
//   lane>=16: add +8 to K                                   (K 8..15,24..31)
// Fragment (m,kb) = 512 halves (1 KB), lane-major, 32B per lane (coalesced).
// ---------------------------------------------------------------------------
__global__ void repack_weights(const float* __restrict__ w2,
                               const float* __restrict__ w3,
                               __bf16* __restrict__ w2A,
                               __bf16* __restrict__ w3A)
{
    const int idx = blockIdx.x * 256 + threadIdx.x;
    const int L2H = 16 * KB2 * 512;              // 98304 halves
    const int L3H = 16 * KB3 * 512;              // 196608 halves
    if (idx < L2H) {
        int frag = idx >> 9, within = idx & 511;
        int lane = within >> 4, t = within & 15;
        int m = frag / KB2, kb = frag % KB2;
        int K = kb * 32 + (t < 8 ? t : t + 8) + (lane >= 16 ? 8 : 0);
        int M = m * 16 + (lane & 15);
        int ci = K & 127, k = K >> 7;
        w2A[idx] = (__bf16)w2[M * 384 + ci * 3 + k];
    } else if (idx < L2H + L3H) {
        int j = idx - L2H;
        int frag = j >> 9, within = j & 511;
        int lane = within >> 4, t = within & 15;
        int m = frag / KB3, kb = frag % KB3;
        int K = kb * 32 + (t < 8 ? t : t + 8) + (lane >= 16 ? 8 : 0);
        int M = m * 16 + (lane & 15);
        int ci = K & 255, k = K >> 8;
        w3A[j] = (__bf16)w3[M * 768 + ci * 3 + k];
    }
}

// Fold conv bias + BN into per-channel scale/shift: y = conv*scale + shift.
__global__ void bn_prep(const float* b1, const float* g1, const float* be1,
                        const float* m1, const float* v1,
                        const float* b2, const float* g2, const float* be2,
                        const float* m2, const float* v2,
                        const float* b3, const float* g3, const float* be3,
                        const float* m3, const float* v3,
                        float* __restrict__ bn)
{
    int i = blockIdx.x * 256 + threadIdx.x;
    if (i < 128) {
        float inv = g1[i] / sqrtf(v1[i] + 1e-5f);
        bn[i] = inv;             bn[128 + i]  = be1[i] + (b1[i] - m1[i]) * inv;
    } else if (i < 384) {
        int c = i - 128;
        float inv = g2[c] / sqrtf(v2[c] + 1e-5f);
        bn[256 + c] = inv;       bn[512 + c]  = be2[c] + (b2[c] - m2[c]) * inv;
    } else if (i < 640) {
        int c = i - 384;
        float inv = g3[c] / sqrtf(v3[c] + 1e-5f);
        bn[768 + c] = inv;       bn[1024 + c] = be3[c] + (b3[c] - m3[c]) * inv;
    }
}

// Two ds_load_b128 composed in registers (no alloca / no union).
__device__ __forceinline__ v16bf lds_b_frag(const __bf16* p) {
    v4u lo = *(const v4u*)p;            // addresses are 16B-aligned
    v4u hi = *(const v4u*)(p + 16);
    v8u w  = __builtin_shufflevector(lo, hi, 0, 1, 2, 3, 4, 5, 6, 7);
    return __builtin_bit_cast(v16bf, w);
}

// Load all 7 B fragments first (clause-able ds_load_b128 burst), then issue
// the 7 WMMAs: the scheduler can retire LDS latency with partial dscnt waits
// while earlier WMMAs execute.
template <int STRIDE>
__device__ __forceinline__ void wmma_group7(v16bf a, const __bf16* bp, v8f (&acc)[NT]) {
    v16bf bf[NT];
    #pragma unroll
    for (int t = 0; t < NT; ++t) bf[t] = lds_b_frag(bp + (t * 16) * STRIDE);
    #pragma unroll
    for (int t = 0; t < NT; ++t)
        acc[t] = __builtin_amdgcn_wmma_f32_16x16x32_bf16(
            false, a, false, bf[t], (short)0, acc[t], false, false);
}

__global__ __launch_bounds__(256)
void fused_conv_stack(const float* __restrict__ x,
                      const float* __restrict__ w1,
                      const __bf16* __restrict__ w2A,
                      const __bf16* __restrict__ w3A,
                      const float* __restrict__ bn,
                      float* __restrict__ out)
{
    __shared__ __align__(16) __bf16 h1[H1ROWS * H1S];   // 31,008 B
    __shared__ __align__(16) __bf16 h2[H2ROWS * H2S];   // 60,192 B
    __shared__ float xs[112];
    __shared__ float sums[C3];

    const int n    = blockIdx.x;          // row 0..1023
    const int tid  = threadIdx.x;
    const int lane = tid & 31;
    const int wv   = tid >> 5;            // 8 waves
    const int ln   = lane & 15;
    const bool hi  = lane >= 16;
    const int kup  = hi ? 8 : 0;          // B layout: upper lanes take K+8 (halves 8..15 at K+16)

    const float* scale1 = bn;        const float* shift1 = bn + 128;
    const float* scale2 = bn + 256;  const float* shift2 = bn + 512;
    const float* scale3 = bn + 768;  const float* shift3 = bn + 1024;

    if (tid < C3) sums[tid] = 0.0f;
    // h2 rows 112..113 are read (conv halo) but never written by layer2: zero once.
    for (int i = tid; i < 2 * H2S; i += 256) h2[112 * H2S + i] = (__bf16)0.0f;

    for (int c = 0; c < NCHUNK; ++c) {
        const int base = c * CHUNK;
        __syncthreads();

        // ---- stage x halo: xs[i] = x[base-3+i], zero-padded ----
        for (int i = tid; i < 112; i += 256) {
            int p = base - 3 + i;
            xs[i] = (p >= 0 && p < L_TOTAL) ? x[n * L_TOTAL + p] : 0.0f;
        }
        __syncthreads();

        // ---- layer 1 (scalar, tiny): h1T[j][ci], position p1 = base-2+j ----
        for (int idx = tid; idx < H1ROWS * C1; idx += 256) {
            int j = idx >> 7, ci = idx & 127;
            float v = 0.0f;
            int p1 = base - 2 + j;
            if (j <= CHUNK + 3 && p1 >= 0 && p1 < L_TOTAL) {
                float s = w1[ci * 3 + 0] * xs[j]
                        + w1[ci * 3 + 1] * xs[j + 1]
                        + w1[ci * 3 + 2] * xs[j + 2];
                v = fmaxf(fmaf(s, scale1[ci], shift1[ci]), 0.0f);
            }
            h1[j * H1S + ci] = (__bf16)v;
        }
        __syncthreads();

        // ---- layer 2: WMMA GEMM, D[co][j2] = sum_K w2A[co][K] * h1T[j2+k][ci] ----
        #pragma clang loop unroll(disable)
        for (int mi = 0; mi < 2; ++mi) {
            const int m = wv * 2 + mi;                 // 16 M-tiles / 8 waves
            v8f acc[NT] = {};
            const __bf16* abase = w2A + m * (KB2 * 512) + lane * 16;
            v16bf a0 = *(const v16bf*)abase;
            // 2x ping-pong: A prefetch after previous A's last use (no copies).
            // Final (kb+2) prefetch reads one fragment past this m's block —
            // still inside d_ws (next m / w3A / bn), intentionally unconditional.
            #pragma clang loop unroll(disable)
            for (int kb = 0; kb < KB2; kb += 2) {
                v16bf a1 = *(const v16bf*)(abase + (kb + 1) * 512);
                {
                    int K0 = kb * 32 + kup, k = K0 >> 7, cb = K0 & 127;
                    wmma_group7<H1S>(a0, &h1[(ln + k) * H1S + cb], acc);
                }
                a0 = *(const v16bf*)(abase + (kb + 2) * 512);
                {
                    int K0 = (kb + 1) * 32 + kup, k = K0 >> 7, cb = K0 & 127;
                    wmma_group7<H1S>(a1, &h1[(ln + k) * H1S + cb], acc);
                }
            }
            // BN + ReLU + SAME-pad masking, store bf16 h2T (r outer: hoist BN consts)
            #pragma unroll
            for (int r = 0; r < 8; ++r) {
                int co = m * 16 + r + (hi ? 8 : 0);    // C/D layout: hi lanes M+8
                float sc = scale2[co], sh = shift2[co];
                #pragma unroll
                for (int t = 0; t < NT; ++t) {
                    int j2 = t * 16 + ln;
                    int p2 = base - 1 + j2;
                    bool valid = (j2 <= CHUNK + 1) && (p2 >= 0) && (p2 < L_TOTAL);
                    float v = valid ? fmaxf(fmaf(acc[t][r], sc, sh), 0.0f) : 0.0f;
                    h2[j2 * H2S + co] = (__bf16)v;
                }
            }
        }
        __syncthreads();

        // ---- layer 3: WMMA GEMM + BN/ReLU + masked mean accumulation ----
        #pragma clang loop unroll(disable)
        for (int mi = 0; mi < 2; ++mi) {
            const int m = wv * 2 + mi;
            v8f acc[NT] = {};
            const __bf16* abase = w3A + m * (KB3 * 512) + lane * 16;
            v16bf a0 = *(const v16bf*)abase;
            #pragma clang loop unroll(disable)
            for (int kb = 0; kb < KB3; kb += 2) {
                v16bf a1 = *(const v16bf*)(abase + (kb + 1) * 512);
                {
                    int K0 = kb * 32 + kup, k = K0 >> 8, cb = K0 & 255;
                    wmma_group7<H2S>(a0, &h2[(ln + k) * H2S + cb], acc);
                }
                a0 = *(const v16bf*)(abase + (kb + 2) * 512);
                {
                    int K0 = (kb + 1) * 32 + kup, k = K0 >> 8, cb = K0 & 255;
                    wmma_group7<H2S>(a1, &h2[(ln + k) * H2S + cb], acc);
                }
            }
            #pragma unroll
            for (int r = 0; r < 8; ++r) {
                int co = m * 16 + r + (hi ? 8 : 0);
                float sc = scale3[co], sh = shift3[co];
                float ps = 0.0f;
                #pragma unroll
                for (int t = 0; t < NT; ++t) {
                    int li = t * 16 + ln;                  // output pos base+li
                    if (li < CHUNK)
                        ps += fmaxf(fmaf(acc[t][r], sc, sh), 0.0f);
                }
                atomicAdd(&sums[co], ps);                  // ds_add_f32
            }
        }
    }
    __syncthreads();
    if (tid < C3) out[n * C3 + tid] = sums[tid] * (1.0f / (float)L_TOTAL);
}

// ---------------------------------------------------------------------------
// Launcher. Workspace layout (needs ~595 KB):
//   [0)            w2A  : 16*12*512 bf16 = 196,608 B
//   [196608)       w3A  : 16*24*512 bf16 = 393,216 B
//   [589824)       bn   : 1280 f32       =   5,120 B  (also prefetch slack)
// ---------------------------------------------------------------------------
extern "C" void kernel_launch(void* const* d_in, const int* in_sizes, int n_in,
                              void* d_out, int out_size, void* d_ws, size_t ws_size,
                              hipStream_t stream) {
    const float* x   = (const float*)d_in[0];
    const float* w1  = (const float*)d_in[1];
    const float* b1  = (const float*)d_in[2];
    const float* w2  = (const float*)d_in[3];
    const float* b2  = (const float*)d_in[4];
    const float* w3  = (const float*)d_in[5];
    const float* b3  = (const float*)d_in[6];
    const float* g1  = (const float*)d_in[7];
    const float* be1 = (const float*)d_in[8];
    const float* m1  = (const float*)d_in[9];
    const float* v1  = (const float*)d_in[10];
    const float* g2  = (const float*)d_in[11];
    const float* be2 = (const float*)d_in[12];
    const float* m2  = (const float*)d_in[13];
    const float* v2  = (const float*)d_in[14];
    const float* g3  = (const float*)d_in[15];
    const float* be3 = (const float*)d_in[16];
    const float* m3  = (const float*)d_in[17];
    const float* v3  = (const float*)d_in[18];

    __bf16* w2A = (__bf16*)d_ws;
    __bf16* w3A = w2A + 16 * KB2 * 512;
    float*  bn  = (float*)((char*)d_ws + 589824);

    repack_weights<<<1152, 256, 0, stream>>>(w2, w3, w2A, w3A);
    bn_prep<<<3, 256, 0, stream>>>(b1, g1, be1, m1, v1,
                                   b2, g2, be2, m2, v2,
                                   b3, g3, be3, m3, v3, bn);
    fused_conv_stack<<<1024, 256, 0, stream>>>(x, w1, w2A, w3A, bn, (float*)d_out);
}
